// Cache_33603824124053
// MI455X (gfx1250) — matrix operand order repaired and checked by reading, not compiled, not run on hardware
//
#include <hip/hip_runtime.h>
#include <hip/hip_bf16.h>

// ---------------------------------------------------------------------------
// Problem constants (from the reference setup)
// ---------------------------------------------------------------------------
#define QLEN   64
#define LSTEPS 128
#define BATCH  32
#define HID    256
#define K_TOT  (LSTEPS * HID)      // 32768  (GEMM K)
#define DKDIM  64                  // summary dim / N of GEMM
#define NSLOT  64                  // cache slots
#define MROWS  (QLEN * BATCH)      // 2048   (GEMM M)
#define TOPK   4
#define KWAVES 16                  // K-split waves per block
#define KSLICE (K_TOT / KWAVES)    // 2048

typedef __attribute__((ext_vector_type(16))) __bf16 v16bf;
typedef __attribute__((ext_vector_type(4)))  __bf16 v4bf;
typedef __attribute__((ext_vector_type(8)))  float  v8f;

// Split one fp32 value into bf16 hi + bf16 lo (hi+lo captures 16 mantissa bits,
// giving ~2^-17 relative error after the 3-product reconstruction).
#define SPLIT1(HH, LL, I, X)                          \
  do {                                                \
    float  _x = (X);                                  \
    __bf16 _h = (__bf16)_x;                           \
    (HH)[(I)] = _h;                                   \
    (LL)[(I)] = (__bf16)(_x - (float)_h);             \
  } while (0)

__device__ inline void split4(v16bf& hh, v16bf& ll, int i, float4 v) {
  SPLIT1(hh, ll, i + 0, v.x);
  SPLIT1(hh, ll, i + 1, v.y);
  SPLIT1(hh, ll, i + 2, v.z);
  SPLIT1(hh, ll, i + 3, v.w);
}

// ---------------------------------------------------------------------------
// Pre-pass: split W_summary (fp32 [64][32768]) once into bf16 hi/lo arrays so
// the GEMM's B fragments need no per-wave conversion VALU and half the bytes
// per operand load. 2048 blocks x 256 threads x 4 elems = 2,097,152 elems.
// ---------------------------------------------------------------------------
__global__ __launch_bounds__(256) void presplit_W(
    const float* __restrict__ W, __bf16* __restrict__ Whi,
    __bf16* __restrict__ Wlo) {
  const size_t i = ((size_t)blockIdx.x * 256 + threadIdx.x) * 4;
  float4 v = *(const float4*)(W + i);
  v4bf hh, ll;
  SPLIT1(hh, ll, 0, v.x);
  SPLIT1(hh, ll, 1, v.y);
  SPLIT1(hh, ll, 2, v.z);
  SPLIT1(hh, ll, 3, v.w);
  *(v4bf*)(Whi + i) = hh;
  *(v4bf*)(Wlo + i) = ll;
}

// ---------------------------------------------------------------------------
// Stage 1: summary GEMM  qsum[m,d] = 0.125*(sum_k A[m,k]*W[d,k] + b[d])
//   A[m,k] = query[q,l,b,h], m=q*32+b, k=l*256+h (k-contiguous in h).
//   bf16x3 (hi*hi + hi*lo + lo*hi) via v_wmma_f32_16x16x32_bf16.
//   Block = one q (32 rows, two 16-row sub-tiles) x all N=64;
//   16 waves K-split (2048 K each); B fragments reused across both sub-tiles;
//   cross-wave reduction through 128 KB LDS.
// ---------------------------------------------------------------------------
template <bool PRESPLIT>
__global__ __launch_bounds__(512) void summary_gemm(
    const float* __restrict__ query, const float* __restrict__ Wf32,
    const __bf16* __restrict__ Whi, const __bf16* __restrict__ Wlo,
    const float* __restrict__ bsum, float* __restrict__ qsum) {
  const int lane = threadIdx.x & 31;
  const int kw   = threadIdx.x >> 5;       // K-slice id, 0..15
  const int qidx = blockIdx.x;             // one q per block (M-tile of 32)
  const int mrow = lane & 15;              // row within 16-row sub-tile
  const int sel  = lane >> 4;              // which K-half this lane packs

  // element offset of A[q, l=0, b, h=0] + this lane's sel*16 sub-offset;
  // sub-tile 1 (b += 16) is a constant +16*HID away.
  const size_t rowbase = (size_t)qidx * (size_t)(LSTEPS * BATCH * HID) +
                         (size_t)mrow * HID + (size_t)(sel * 16);

  v8f acc[2][4] = {};                      // 32x64 fp32 accumulators

  const int kbeg = kw * KSLICE;
  const int kend = kbeg + KSLICE;
  for (int kk = kbeg; kk < kend; kk += 32) {
    const int l  = kk >> 8;                // 32 | 256: k-step stays in one l
    const int h0 = kk & 255;
    const float* pa0 = query + rowbase + (size_t)l * (BATCH * HID) + h0;
    const float* pa1 = pa0 + 16 * HID;     // second sub-tile (b+16)
    __builtin_prefetch(pa0 + 128, 0, 3);   // global_prefetch_b8 on A stream
    __builtin_prefetch(pa1 + 128, 0, 3);

    // ---- A fragments: contiguous K order matching B ---------------------
    // lane<16: K = {0..15}; lane>=16: K = {16..31}
    v16bf Ah[2], Al[2];
    split4(Ah[0], Al[0], 0,  *(const float4*)(pa0 + 0));
    split4(Ah[0], Al[0], 4,  *(const float4*)(pa0 + 4));
    split4(Ah[0], Al[0], 8,  *(const float4*)(pa0 + 8));
    split4(Ah[0], Al[0], 12, *(const float4*)(pa0 + 12));
    split4(Ah[1], Al[1], 0,  *(const float4*)(pa1 + 0));
    split4(Ah[1], Al[1], 4,  *(const float4*)(pa1 + 4));
    split4(Ah[1], Al[1], 8,  *(const float4*)(pa1 + 8));
    split4(Ah[1], Al[1], 12, *(const float4*)(pa1 + 12));

    // ---- B fragments per 16-col N tile, shared by both sub-tiles -------
#pragma unroll
    for (int nt = 0; nt < 4; ++nt) {
      const int col = nt * 16 + (lane & 15);        // output column (dk)
      const size_t boff = (size_t)col * K_TOT + kk + sel * 16;
      v16bf Bh, Bl;
      if constexpr (PRESPLIT) {
        Bh = *(const v16bf*)(Whi + boff);           // 32B aligned
        Bl = *(const v16bf*)(Wlo + boff);
      } else {
        const float* pb = Wf32 + boff;
        split4(Bh, Bl, 0,  *(const float4*)(pb + 0));
        split4(Bh, Bl, 4,  *(const float4*)(pb + 4));
        split4(Bh, Bl, 8,  *(const float4*)(pb + 8));
        split4(Bh, Bl, 12, *(const float4*)(pb + 12));
      }
#pragma unroll
      for (int s = 0; s < 2; ++s) {
        acc[s][nt] = __builtin_amdgcn_wmma_f32_16x16x32_bf16(
            false, Ah[s], false, Bh, (short)0, acc[s][nt], false, false);
        acc[s][nt] = __builtin_amdgcn_wmma_f32_16x16x32_bf16(
            false, Ah[s], false, Bl, (short)0, acc[s][nt], false, false);
        acc[s][nt] = __builtin_amdgcn_wmma_f32_16x16x32_bf16(
            false, Al[s], false, Bh, (short)0, acc[s][nt], false, false);
      }
    }
  }

  // ---- cross-wave K reduction through LDS ------------------------------
  __shared__ float red[KWAVES][32][NSLOT];          // 128 KB
#pragma unroll
  for (int s = 0; s < 2; ++s)
#pragma unroll
    for (int nt = 0; nt < 4; ++nt)
#pragma unroll
      for (int v = 0; v < 8; ++v)  // C/D: elem v -> row v + 8*sel (in subtile)
        red[kw][s * 16 + v + sel * 8][nt * 16 + (lane & 15)] = acc[s][nt][v];
  __syncthreads();

  for (int e = threadIdx.x; e < 32 * NSLOT; e += 512) {
    const int row = e >> 6, col = e & 63;           // row = b, col = d
    float s = 0.f;
#pragma unroll
    for (int w = 0; w < KWAVES; ++w) s += red[w][row][col];
    // fold bias and the 1/sqrt(dk)=1/8 attention scale into qsum
    qsum[((size_t)qidx * BATCH + row) * DKDIM + col] = 0.125f * (s + bsum[col]);
  }
}

// ---------------------------------------------------------------------------
// Stage 2: scores = qsum . keys, softmax over 64 slots, top-4 (ties -> lowest
// index, matching jax.lax.top_k), re-softmax of the selected probabilities.
// One wave per (q,b) row; each lane owns slots {lane, lane+32}.
// ---------------------------------------------------------------------------
__global__ __launch_bounds__(256) void attn_topk(
    const float* __restrict__ qsum, const float* __restrict__ keys,
    float* __restrict__ out_w, float* __restrict__ out_i) {
  const int lane = threadIdx.x & 31;
  const int wv   = threadIdx.x >> 5;
  const int r    = blockIdx.x * 8 + wv;          // row in [0,2048)
  const int b    = r & 31;
  const float* qrow = qsum + (size_t)r * DKDIM;  // uniform per wave

  float s[2];
#pragma unroll
  for (int t = 0; t < 2; ++t) {
    const int n = lane + t * 32;
    const float* krow = keys + ((size_t)n * BATCH + b) * DKDIM;
    float a = 0.f;
#pragma unroll
    for (int d = 0; d < DKDIM; d += 4) {
      float4 kv = *(const float4*)(krow + d);
      float4 qv = *(const float4*)(qrow + d);
      a = fmaf(qv.x, kv.x, a);
      a = fmaf(qv.y, kv.y, a);
      a = fmaf(qv.z, kv.z, a);
      a = fmaf(qv.w, kv.w, a);
    }
    s[t] = a;                                    // qsum pre-scaled by 1/8
  }

  // softmax over the 64 slots (wave32 butterfly reductions)
  float mx = fmaxf(s[0], s[1]);
  for (int off = 16; off > 0; off >>= 1) mx = fmaxf(mx, __shfl_xor(mx, off, 32));
  float e0 = __expf(s[0] - mx), e1 = __expf(s[1] - mx);
  float sum = e0 + e1;
  for (int off = 16; off > 0; off >>= 1) sum += __shfl_xor(sum, off, 32);
  float c0 = e0 / sum, c1 = e1 / sum;            // attn probabilities

  // iterative argmax x4 (descending value, ties -> lower slot index)
  float wsel[TOPK];
  int   isel[TOPK];
#pragma unroll
  for (int t = 0; t < TOPK; ++t) {
    float v;
    int   i;
    if (c0 >= c1) { v = c0; i = lane; }          // lane < lane+32 on ties
    else          { v = c1; i = lane + 32; }
    for (int off = 16; off > 0; off >>= 1) {
      float ov = __shfl_xor(v, off, 32);
      int   oi = __shfl_xor(i, off, 32);
      if (ov > v || (ov == v && oi < i)) { v = ov; i = oi; }
    }
    wsel[t] = v;
    isel[t] = i;
    if (i == lane)      c0 = -1.f;               // remove winner
    if (i == lane + 32) c1 = -1.f;
  }

  // re-softmax of the 4 selected probabilities (wsel[0] is the max)
  float ee[TOPK], s2 = 0.f;
#pragma unroll
  for (int t = 0; t < TOPK; ++t) { ee[t] = __expf(wsel[t] - wsel[0]); s2 += ee[t]; }

  if (lane == 0) {
#pragma unroll
    for (int t = 0; t < TOPK; ++t) {
      out_w[(size_t)r * TOPK + t] = ee[t] / s2;         // [2048,1,4] flat
      out_i[(size_t)t * MROWS + r] = (float)isel[t];    // [4,2048] flat (exact)
    }
  }
}

// ---------------------------------------------------------------------------
// Launch. Inputs: query, W_summary, b_summary, keys, values(unused by outputs).
// d_out = topk_weights (2048*4) ++ topk_indices (4*2048) in return order.
// ws layout: qsum fp32 [2048*64] | Whi bf16 [64*32768] | Wlo bf16 [64*32768]
// ---------------------------------------------------------------------------
extern "C" void kernel_launch(void* const* d_in, const int* in_sizes, int n_in,
                              void* d_out, int out_size, void* d_ws,
                              size_t ws_size, hipStream_t stream) {
  const float* query = (const float*)d_in[0];
  const float* Wsum  = (const float*)d_in[1];
  const float* bsum  = (const float*)d_in[2];
  const float* keys  = (const float*)d_in[3];
  // d_in[4] (values) is dead: the reference discards the attention output.

  const size_t qsum_bytes = (size_t)MROWS * DKDIM * sizeof(float);   // 512 KB
  const size_t wspl_elems = (size_t)DKDIM * K_TOT;                   // 2 M
  const size_t need = qsum_bytes + 2 * wspl_elems * sizeof(__bf16);  // ~8.5 MB

  float*  qsum  = (float*)d_ws;
  float*  out_w = (float*)d_out;                        // weights [2048,1,4]
  float*  out_i = (float*)d_out + (size_t)MROWS * TOPK; // indices [4,2048]

  if (ws_size >= need) {
    __bf16* Whi = (__bf16*)((char*)d_ws + qsum_bytes);
    __bf16* Wlo = Whi + wspl_elems;
    presplit_W<<<dim3((unsigned)(wspl_elems / 1024)), dim3(256), 0, stream>>>(
        Wsum, Whi, Wlo);
    summary_gemm<true><<<dim3(QLEN), dim3(512), 0, stream>>>(
        query, Wsum, Whi, Wlo, bsum, qsum);
  } else {
    summary_gemm<false><<<dim3(QLEN), dim3(512), 0, stream>>>(
        query, Wsum, (const __bf16*)nullptr, (const __bf16*)nullptr, bsum,
        qsum);
  }

  attn_topk<<<dim3(MROWS / 8), dim3(256), 0, stream>>>(qsum, keys, out_w,
                                                       out_i);
}
